// Kerception_blockC_28621662060934
// MI455X (gfx1250) — matrix operand order repaired and checked by reading, not compiled, not run on hardware
//
#include <hip/hip_runtime.h>
#include <math.h>

// ---------------------------------------------------------------------------
// Kerception block for MI455X (gfx1250, wave32, WMMA).
//
// GEMM: M=8*224*224 pixels, N=16 outputs, K=576 (3x3x64 patch), f16 inputs /
// f32 accumulate via v_wmma_f32_16x16x32_f16. One workgroup per (batch,row):
// 14 waves x 16 pixels = 224 columns; x rows h-1..h+1 staged to LDS as f16.
// HBM floor ~5.5us (103MB in + 26MB out @ 23.3TB/s); halo re-reads hit the
// 192MB L2. Compute on f16 WMMA is ~1-2us -> memory bound, as intended.
// ---------------------------------------------------------------------------

typedef __attribute__((ext_vector_type(16))) _Float16 v16h;
typedef __attribute__((ext_vector_type(8)))  _Float16 v8h;
typedef __attribute__((ext_vector_type(8)))  float    v8f;
typedef __attribute__((ext_vector_type(4)))  float    v4f;

#define BB 8
#define HH 224
#define WW 224
#define CC 64
#define DD 576           // K = 9*64
#define NN 16            // concat output width: 1+1+4+5+5
#define PW 226           // padded row width (halo of 1 each side)
#define NTHREADS 448     // 14 waves * 32

// dynamic-LDS partition (bytes)
#define XT_OFF  0                       // _Float16 xt[3][PW][CC]  = 86784 B
#define WT_OFF  86784                   // _Float16 wt[NN][DD]     = 18432 B
#define SQ_OFF  105216                  // float    sq[3][PW]      =  2712 B
#define WW_OFF  107928                  // float    wwv[4]         =    16 B
#define LDS_TOTAL 107944

extern "C" __global__ void __launch_bounds__(NTHREADS)
kerception_wmma_kernel(const float* __restrict__ x,
                       const float* __restrict__ w1,
                       const float* __restrict__ w2,
                       const float* __restrict__ w3,
                       const float* __restrict__ w4,
                       const float* __restrict__ w5,
                       const float* __restrict__ gamma_p,
                       const float* __restrict__ c4_p,
                       const float* __restrict__ c5_p,
                       float* __restrict__ out)
{
    extern __shared__ char smem[];
    _Float16* xt  = (_Float16*)(smem + XT_OFF);   // [3][PW][CC] f16 tile
    _Float16* wt  = (_Float16*)(smem + WT_OFF);   // [NN][DD]    f16 weights^T (k-permuted)
    float*    sq  = (float*)   (smem + SQ_OFF);   // [3][PW]     sum_c x^2
    float*    wwv = (float*)   (smem + WW_OFF);   // [4]         colsum(w3^2)

    const int tid = threadIdx.x;
    const int b   = blockIdx.x / HH;
    const int h   = blockIdx.x % HH;

    // ---- stage x rows h-1..h+1 -> LDS as f16, zero padding at borders -----
    for (int idx = tid; idx < 3 * PW * (CC / 4); idx += NTHREADS) {
        const int c4  = idx % (CC / 4);
        const int t   = idx / (CC / 4);
        const int col = t % PW;
        const int r   = t / PW;
        const int gh  = h + r - 1;
        const int gw  = col - 1;
        v4f v = {0.f, 0.f, 0.f, 0.f};
        if ((unsigned)gh < (unsigned)HH && (unsigned)gw < (unsigned)WW) {
            const float* gp = x + (((size_t)(b * HH + gh) * WW + gw) * CC + c4 * 4);
            v = *(const v4f*)gp;
        }
        _Float16* dp = xt + ((r * PW + col) * CC + c4 * 4);
        dp[0] = (_Float16)v[0]; dp[1] = (_Float16)v[1];
        dp[2] = (_Float16)v[2]; dp[3] = (_Float16)v[3];
    }

    // ---- stage permuted weights: our K-order k = s*64 + c  (d = c*9 + s) --
    // columns: [w1 | w2 | w3(4) | w4(5) | w5(5)]
    for (int idx = tid; idx < NN * DD; idx += NTHREADS) {
        const int n = idx / DD;
        const int k = idx % DD;
        const int s = k >> 6;        // spatial offset 0..8 (ky*3+kx)
        const int c = k & 63;        // channel
        const int d = c * 9 + s;     // original patch index
        float w;
        if      (n == 0) w = w1[d];
        else if (n == 1) w = w2[d];
        else if (n <  6) w = w3[d * 4 + (n - 2)];
        else if (n < 11) w = w4[d * 5 + (n - 6)];
        else             w = w5[d * 5 + (n - 11)];
        wt[n * DD + k] = (_Float16)w;
    }
    if (tid < 4) {  // ww[j] = sum_d w3[d][j]^2  (fp32)
        float a = 0.f;
        for (int d = 0; d < DD; ++d) { const float w = w3[d * 4 + tid]; a += w * w; }
        wwv[tid] = a;
    }
    __syncthreads();

    // ---- per-position channel sum of squares (for gaussian branch) -------
    for (int idx = tid; idx < 3 * PW; idx += NTHREADS) {
        const _Float16* p = xt + idx * CC;
        float a = 0.f;
        for (int c = 0; c < CC; ++c) { const float v = (float)p[c]; a += v * v; }
        sq[idx] = a;
    }
    __syncthreads();

    // ---- WMMA main loop: one 16px x 16out tile per wave -------------------
    const int lane = tid & 31;
    const int wave = tid >> 5;       // 0..13
    const int hi   = lane >> 4;      // lane group (0: lanes 0-15, 1: 16-31)
    const int ln   = lane & 15;
    const int w0   = wave * 16;      // first pixel column of this tile
    const int mcol = w0 + ln + 1;    // padded col for A row (M = ln)

    v8f acc = {};
    #pragma unroll
    for (int s = 0; s < 9; ++s) {
        const int ky = s / 3;
        const int kx = s % 3;
        const _Float16* arow = xt + ((ky * PW + (mcol + kx - 1)) * CC);
        #pragma unroll
        for (int u = 0; u < 2; ++u) {
            const int kc = s * 2 + u;          // K chunk (32 channels)
            // A 16x32 f16, this code's k order (B matches): halves 0-15 -> K=hi*16+0..15
            const v8h alo = *(const v8h*)(arow + u * 32 + hi * 16);
            const v8h ahi = *(const v8h*)(arow + u * 32 + hi * 16 + 8);
            v16h a;
            #pragma unroll
            for (int i = 0; i < 8; ++i) { a[i] = alo[i]; a[i + 8] = ahi[i]; }
            // B 32x16 f16: lane=column n, halves = K = hi*16 + 0..15 (this code's order, same as A)
            const v16h bf = *(const v16h*)(wt + ln * DD + kc * 32 + hi * 16);
            acc = __builtin_amdgcn_wmma_f32_16x16x32_f16(
                      false, a, false, bf, (short)0, acc, false, false);
        }
    }

    // ---- epilogue: C/D layout lane n=ln, VGPR r -> pixel m = hi*8 + r -----
    const float gamma = *gamma_p;
    const float c4v   = *c4_p;
    const float c5v   = *c5_p;
    const int   n     = ln;
    const float wwn   = (n >= 2 && n < 6) ? wwv[n - 2] : 0.f;

    #pragma unroll
    for (int r = 0; r < 8; ++r) {
        const int   m   = hi * 8 + r;
        const int   col = w0 + m + 1;
        const float y   = acc[r];
        float o;
        if      (n == 0) o = y;
        else if (n == 1) o = tanhf(y);
        else if (n <  6) {
            const float xx = sq[0 * PW + col - 1] + sq[0 * PW + col] + sq[0 * PW + col + 1]
                           + sq[1 * PW + col - 1] + sq[1 * PW + col] + sq[1 * PW + col + 1]
                           + sq[2 * PW + col - 1] + sq[2 * PW + col] + sq[2 * PW + col + 1];
            o = expf(-gamma * (xx - 2.0f * y + wwn));
        } else if (n < 11) {
            const float t = y + c4v; o = t * t * t;
        } else {
            const float t = y + c5v; const float t2 = t * t; o = t2 * t2 * t;
        }
        const int w = w0 + m;
        out[(((size_t)(b * HH + h)) * WW + w) * NN + n] = o;
    }
}

extern "C" void kernel_launch(void* const* d_in, const int* in_sizes, int n_in,
                              void* d_out, int out_size, void* d_ws, size_t ws_size,
                              hipStream_t stream) {
    (void)in_sizes; (void)n_in; (void)d_ws; (void)ws_size; (void)out_size;
    const float* x     = (const float*)d_in[0];
    const float* w1    = (const float*)d_in[1];
    const float* w2    = (const float*)d_in[2];
    const float* w3    = (const float*)d_in[3];
    const float* w4    = (const float*)d_in[4];
    const float* w5    = (const float*)d_in[5];
    const float* gamma = (const float*)d_in[6];
    const float* c4    = (const float*)d_in[7];
    const float* c5    = (const float*)d_in[8];
    float* out = (float*)d_out;

    (void)hipFuncSetAttribute((const void*)kerception_wmma_kernel,
                              hipFuncAttributeMaxDynamicSharedMemorySize,
                              (int)LDS_TOTAL);

    dim3 grid(BB * HH);      // one workgroup per (batch, row) = 1792
    dim3 block(NTHREADS);    // 14 waves, one 16x16 output tile each
    kerception_wmma_kernel<<<grid, block, LDS_TOTAL, stream>>>(
        x, w1, w2, w3, w4, w5, gamma, c4, c5, out);
}
